// TRILoss_3908420240117
// MI455X (gfx1250) — matrix operand order repaired and checked by reading, not compiled, not run on hardware
//
#include <hip/hip_runtime.h>
#include <stdint.h>

typedef __attribute__((ext_vector_type(16))) _Float16 v16h;
typedef __attribute__((ext_vector_type(8)))  float    v8f;

#define B_SZ    4096
#define N_SZ    8192
#define D_SZ    128
#define V_SZ    2
#define NROUNDS 5
#define HALF_CNT (1u << 25)        // (N*N)/2 = 2^26 / 2
#define EPS_C   1e-6f
#define TINY_C  1.17549435e-38f    // FLT_MIN == jnp.finfo(f32).tiny

struct U2 { uint32_t x, y; };

__device__ __forceinline__ uint32_t rotl32(uint32_t v, int r) {
  return (v << r) | (v >> (32 - r));
}

// Exact JAX threefry2x32 block (jax/_src/prng.py)
__device__ __forceinline__ U2 threefry2x32(uint32_t k0, uint32_t k1,
                                           uint32_t c0, uint32_t c1) {
  const uint32_t ks2 = k0 ^ k1 ^ 0x1BD11BDAu;
  uint32_t v0 = c0 + k0, v1 = c1 + k1;
#define TFR(r) { v0 += v1; v1 = rotl32(v1, r); v1 ^= v0; }
  TFR(13) TFR(15) TFR(26) TFR(6)   v0 += k1;  v1 += ks2 + 1u;
  TFR(17) TFR(29) TFR(16) TFR(24)  v0 += ks2; v1 += k0  + 2u;
  TFR(13) TFR(15) TFR(26) TFR(6)   v0 += k0;  v1 += k1  + 3u;
  TFR(17) TFR(29) TFR(16) TFR(24)  v0 += k1;  v1 += ks2 + 4u;
  TFR(13) TFR(15) TFR(26) TFR(6)   v0 += ks2; v1 += k0  + 5u;
#undef TFR
  U2 o; o.x = v0; o.y = v1; return o;
}

// key(42) -> fold_in(k) -> split -> (kp, kn)
__device__ __forceinline__ U2 derive_key(int round, int pn) {
  U2 f = threefry2x32(0u, 42u, 0u, (uint32_t)round);   // fold_in: block(0, k)
  U2 a = threefry2x32(f.x, f.y, 0u, 2u);               // split counts [0,1,2,3]
  U2 b = threefry2x32(f.x, f.y, 1u, 3u);
  U2 r;
  if (pn == 0) { r.x = a.x; r.y = b.x; }               // kp = (a0, b0)
  else         { r.x = a.y; r.y = b.y; }               // kn = (a1, b1)
  return r;
}

// -------- Kernel 1: gumbel-argmax sampling of pos/neg indices --------
// One wave per task = (round, pn, row). task layout: (round*2 + pn)*N + row.
__global__ void __launch_bounds__(256)
trisample_kernel(const int* __restrict__ labels, int* __restrict__ idx_out) {
  __shared__ int slab[B_SZ];
  for (int t = threadIdx.x; t < B_SZ; t += blockDim.x) slab[t] = labels[t];
  __syncthreads();

  const int wave = threadIdx.x >> 5;
  const int lane = threadIdx.x & 31;
  const int task = blockIdx.x * (blockDim.x >> 5) + wave;
  const int round = task >> 14;
  const int pn    = (task >> 13) & 1;
  const int row   = task & (N_SZ - 1);

  const U2 key = derive_key(round, pn);
  const int myLab = slab[row & (B_SZ - 1)];
  const uint32_t rowFlat = (uint32_t)row << 13;        // row * N
  const bool useWord1 = (row >= (N_SZ / 2));           // flat >= 2^25

  float bestG = -__builtin_inff();
  int   bestJ = N_SZ;

  for (int j = lane; j < N_SZ; j += 32) {
    const bool samePos = (slab[j & (B_SZ - 1)] == myLab) && (j != row);
    const bool valid = (pn == 0) ? samePos : !samePos; // neg includes diagonal
    if (!valid) continue;
    const uint32_t flat = rowFlat + (uint32_t)j;
    const uint32_t c0 = useWord1 ? (flat - HALF_CNT) : flat;
    const uint32_t c1 = useWord1 ? flat : (flat + HALF_CNT);
    const U2 rb = threefry2x32(key.x, key.y, c0, c1);
    const uint32_t bits = useWord1 ? rb.y : rb.x;
    float u = __uint_as_float((bits >> 9) | 0x3f800000u) - 1.0f;
    float v = u * (1.0f - TINY_C) + TINY_C;
    v = fmaxf(TINY_C, v);
    const float g = -logf(-logf(v));
    if (g > bestG) { bestG = g; bestJ = j; }           // ascending j: first-max kept
  }
  // cross-lane argmax, smaller index wins ties (jnp.argmax semantics)
  for (int off = 16; off; off >>= 1) {
    const float og = __shfl_xor(bestG, off, 32);
    const int   oj = __shfl_xor(bestJ, off, 32);
    if (og > bestG || (og == bestG && oj < bestJ)) { bestG = og; bestJ = oj; }
  }
  if (lane == 0) idx_out[task] = bestJ;
}

// -------- Kernel 2: WMMA Gram-diagonal distances + hinge reduction --------
// 8 waves/block, 16 rows/wave. dp/dn via diag(X X^T), X = cf[r] - cf[idx[r]] + eps.
__global__ void __launch_bounds__(256)
triloss_kernel(const float* __restrict__ features,
               const int* __restrict__ idx_buf,
               float* __restrict__ out) {
  __shared__ float tile[8 * 16 * D_SZ];                // 64 KB (<< 320 KB/WGP)
  const int wave = threadIdx.x >> 5;
  const int lane = threadIdx.x & 31;
  const int round = blockIdx.y;
  const int rowBase = blockIdx.x * 128 + wave * 16;
  float* X = &tile[wave * 16 * D_SZ];

  float dist[2];
  for (int pn = 0; pn < 2; ++pn) {
    const int* idxArr = idx_buf + (round * 2 + pn) * N_SZ;
    __syncthreads();                                   // tile reuse fence
    // stage 16x128 difference tile (coalesced: 32 lanes = 32 consecutive d)
    for (int e = 0; e < 64; ++e) {
      const int q = e * 32 + lane;
      const int m = q >> 7, d = q & (D_SZ - 1);
      const int r = rowBase + m;
      const int p = idxArr[r];
      const float* fr = features + (((r & (B_SZ - 1)) * V_SZ + (r >> 12)) * D_SZ);
      const float* fp = features + (((p & (B_SZ - 1)) * V_SZ + (p >> 12)) * D_SZ);
      X[q] = fr[d] - fp[d] + EPS_C;
    }
    __syncthreads();

    // Gram = X * X^T via 4x v_wmma_f32_16x16x32_f16 (K = 128)
    v8f acc = {};
    const int m = lane & 15;
#pragma unroll
    for (int c = 0; c < 4; ++c) {
      v16h a, b;
#pragma unroll
      for (int h = 0; h < 16; ++h) {
        // A 16x32 f16 layout: lanes<16 K={0..7,16..23}, lanes>=16 K={8..15,24..31}
        const int da = 32 * c + (h & 7) + ((h < 8) ? 0 : 16) + ((lane < 16) ? 0 : 8);
        a[h] = (_Float16)X[m * D_SZ + da];
        // B 32x16 layout (B = X^T): lanes<16 K={0..7,16..23}, lanes>=16 K={8..15,24..31}, col = lane&15
        const int db = 32 * c + (h & 7) + ((h < 8) ? 0 : 16) + ((lane < 16) ? 0 : 8);
        b[h] = (_Float16)X[m * D_SZ + db];
      }
      acc = __builtin_amdgcn_wmma_f32_16x16x32_f16(
          false, a, false, b, (short)0, acc, false, false);
    }
    // diag(m,m): m<8 -> lane m, VGPR m ; m>=8 -> lane m+16 (24..31), VGPR m-8
    float d2 = 0.0f;
#pragma unroll
    for (int r8 = 0; r8 < 8; ++r8) {
      const int mr = r8 + ((lane >= 16) ? 8 : 0);
      if ((lane & 15) == mr) d2 = acc[r8];
    }
    dist[pn] = sqrtf(d2);
  }

  const bool diagLane = (lane < 8) || (lane >= 24);
  float h = diagLane ? fmaxf(dist[0] - dist[1] + 1.0f, 0.0f) : 0.0f;
  for (int off = 16; off; off >>= 1) h += __shfl_xor(h, off, 32);
  if (lane == 0) atomicAdd(out, h * (1.0f / (float)N_SZ));
}

extern "C" void kernel_launch(void* const* d_in, const int* in_sizes, int n_in,
                              void* d_out, int out_size, void* d_ws, size_t ws_size,
                              hipStream_t stream) {
  const float* features = (const float*)d_in[0];   // [4096, 2, 128] f32
  const int*   labels   = (const int*)d_in[1];     // [4096] i32
  float* out = (float*)d_out;                      // scalar f32 loss
  int* idx_buf = (int*)d_ws;                       // 5*2*8192 ints = 320 KB scratch

  hipMemsetAsync(out, 0, sizeof(float), stream);   // capture-safe zero

  const int nTasks = NROUNDS * 2 * N_SZ;           // 81920 waves
  trisample_kernel<<<nTasks / 8, 256, 0, stream>>>(labels, idx_buf);

  dim3 grid(N_SZ / 128, NROUNDS);                  // 64 x 5 blocks
  triloss_kernel<<<grid, 256, 0, stream>>>(features, idx_buf, out);
}
